// Attention_65180423685192
// MI455X (gfx1250) — matrix operand order repaired and checked by reading, not compiled, not run on hardware
//
#include <hip/hip_runtime.h>
#include <hip/hip_bf16.h>

// ============================================================================
// ViT attention block for MI455X (gfx1250, wave32, WMMA 16x16x32 f16->f32).
//
//   cvt(w_qkv), cvt(w_proj)                  f32 -> f16 weights (L2 resident)
//   qkv_gemm  : x @ w_qkv^T -> Q,K [B,H,208,64] f16 ; V^T [B,H,64,208] f16
//               64-row blocks per WG: each weight (B) fragment is reused by
//               4 WMMAs -> 4x less L2 weight traffic (the roofline limiter).
//   attn      : per (b,h): S = Q K^T / 8, masked softmax (f32, v_exp_f32),
//               P(f16) @ V -> O_f16 [B,208,768]; CLS row -> ga scratch
//   proj_gemm : O @ w_proj^T + b -> d_out[0 .. B*197*768)   (same blocking)
//   ga_reduce : mean over heads  -> d_out[B*197*768 ..)
// ============================================================================

typedef __attribute__((ext_vector_type(16))) _Float16 v16h;
typedef __attribute__((ext_vector_type(8)))  _Float16 v8h;
typedef __attribute__((ext_vector_type(4)))  _Float16 v4h;
typedef __attribute__((ext_vector_type(8)))  float    v8f;

#define WMMA16(a, b, c) \
  __builtin_amdgcn_wmma_f32_16x16x32_f16(false, (a), false, (b), (short)0, (c), false, false)

static constexpr int  B_  = 64;
static constexpr int  N_  = 197;
static constexpr int  NP  = 208;   // padded tokens = 13*16
static constexpr int  C_  = 768;
static constexpr int  H_  = 12;
static constexpr int  HD  = 64;
static constexpr int  PLD = 224;   // P row stride = 7*32 (K-chunks for P@V)

// ---------------------------------------------------------------------------
// WMMA fragment loaders (wave32).
// A (16x32, 16-bit): lane m=lane&15 holds row m; halves 0..15 = K hi*16..+15
//                    (same K order as B)
// B (32x16, 16-bit): lane n=lane&15 holds col n; halves 0..15 = K hi*16..+15
//                    (this code's own K order, not the ISA table order; valid as A matches)
// Both patterns are contiguous-in-K for row-major A and row-major "B^T"
// sources, so q, k (B of q@k^T), w (B of x@w^T) and V^T (B of P@V) all load
// with plain 16B/32B vector loads -- no transpose instructions needed.
// ---------------------------------------------------------------------------
__device__ static inline v16h load_frag_a(const _Float16* base, int ld) {
  const int lane = threadIdx.x & 31;
  const int r = lane & 15, hb = (lane >> 4) * 16;
  return *(const v16h*)(base + r * ld + hb);
}

__device__ static inline v16h load_frag_b(const _Float16* base, int ld) {
  const int lane = threadIdx.x & 31;
  const int r = lane & 15, hb = (lane >> 4) * 16;
  return *(const v16h*)(base + r * ld + hb);
}

// ---------------------------------------------------------------------------
// f32 -> f16 conversion (weights)
// ---------------------------------------------------------------------------
__global__ void cvt_f32_f16(const float* __restrict__ src, _Float16* __restrict__ dst, int n) {
  int i = blockIdx.x * blockDim.x + threadIdx.x;
  int stride = gridDim.x * blockDim.x;
  for (; i < n; i += stride) dst[i] = (_Float16)src[i];
}

// ---------------------------------------------------------------------------
// QKV GEMM: qkv[b,n,d] = sum_c x[b,n,c] * w_qkv[d,c]
// grid: 208 WGs, each owns 4 consecutive 16-row tiles (64 rows) of the
// (B*208)-row padded M space; 256 threads = 8 waves, 18 col-tiles per wave.
// Inner loop: 1 weight B-fragment feeds 4 WMMAs (4 row strips); A-fragments
// preloaded into distinct registers so all 8 ds_loads issue before the waits.
// ---------------------------------------------------------------------------
__global__ __launch_bounds__(256) void qkv_gemm(const float* __restrict__ x,
                                                const _Float16* __restrict__ wq,
                                                _Float16* __restrict__ Q,
                                                _Float16* __restrict__ K,
                                                _Float16* __restrict__ VT) {
  extern __shared__ __align__(32) char smem[];
  _Float16* sA = (_Float16*)smem;              // [4][16][768] = 96 KB
  const int g   = blockIdx.x;                  // 4*g .. 4*g+3 row tiles
  const int tid = threadIdx.x;

  // stage 4 A strips (f32 -> f16), zero-padding token rows >= 197
  for (int i = tid * 4; i < 4 * 16 * C_; i += 256 * 4) {
    const int s  = i / (16 * C_);
    const int j  = i - s * (16 * C_);
    const int rr = j / C_, cc = j - (j / C_) * C_;
    const int rt = 4 * g + s;
    const int b  = rt / 13, bt = rt - (rt / 13) * 13;
    const int n  = bt * 16 + rr;
    v4h o = {};
    if (n < N_) {
      const float4 f = *(const float4*)(x + ((size_t)b * N_ + n) * C_ + cc);
      o[0] = (_Float16)f.x; o[1] = (_Float16)f.y;
      o[2] = (_Float16)f.z; o[3] = (_Float16)f.w;
    }
    *(v4h*)(sA + i) = o;
  }
  __syncthreads();

  const int w    = tid >> 5;
  const int lane = tid & 31;
  const int r    = lane & 15, hi = lane >> 4;

  for (int ct = 0; ct < 18; ++ct) {
    const int ctile = w * 18 + ct;             // 0..143 -> 16 output cols each
    const _Float16* wqt = wq + (size_t)(ctile * 16) * C_;
    __builtin_prefetch(wqt + (size_t)16 * C_, 0, 1);  // next weight tile
    v8f acc[4] = {};
#pragma unroll 2
    for (int kc = 0; kc < 24; ++kc) {          // K = 768 = 24 * 32
      v16h bb = load_frag_b(wqt + kc * 32, C_);
      v16h a0 = load_frag_a(sA + 0 * 16 * C_ + kc * 32, C_);
      v16h a1 = load_frag_a(sA + 1 * 16 * C_ + kc * 32, C_);
      v16h a2 = load_frag_a(sA + 2 * 16 * C_ + kc * 32, C_);
      v16h a3 = load_frag_a(sA + 3 * 16 * C_ + kc * 32, C_);
      acc[0] = WMMA16(a0, bb, acc[0]);
      acc[1] = WMMA16(a1, bb, acc[1]);
      acc[2] = WMMA16(a2, bb, acc[2]);
      acc[3] = WMMA16(a3, bb, acc[3]);
    }
    // scatter: d = t*768 + h*64 + e  (tile never crosses t/h boundary)
    const int d0 = ctile * 16;
    const int t  = d0 / C_;
    const int hh = (d0 % C_) / HD;
    const int e  = (d0 % HD) + r;
#pragma unroll
    for (int s = 0; s < 4; ++s) {
      const int rt = 4 * g + s;
      const int b  = rt / 13, bt = rt - (rt / 13) * 13;
      const size_t bh = (size_t)(b * H_ + hh);
#pragma unroll
      for (int c = 0; c < 8; ++c) {
        const int n = bt * 16 + c + 8 * hi;
        const _Float16 v = (_Float16)acc[s][c];
        if (t == 0)      Q[(bh * NP + n) * HD + e] = v;
        else if (t == 1) K[(bh * NP + n) * HD + e] = v;
        else             VT[(bh * HD + e) * NP + n] = v;   // transposed V
      }
    }
  }
}

// ---------------------------------------------------------------------------
// Attention: one workgroup per (b,h); 13 waves; wave w owns q rows 16w..16w+15.
// Dynamic LDS: K-tile [208][64], V^T [64][224] (zero tail), P [13][16][224].
// ---------------------------------------------------------------------------
__global__ __launch_bounds__(416) void attn_kernel(const _Float16* __restrict__ Q,
                                                   const _Float16* __restrict__ K,
                                                   const _Float16* __restrict__ VT,
                                                   _Float16* __restrict__ O,
                                                   float* __restrict__ ga) {
  extern __shared__ __align__(32) char smem[];
  _Float16* sK = (_Float16*)smem;              // [208][64]
  _Float16* sV = sK + NP * HD;                 // [64][224]
  _Float16* sP = sV + HD * PLD;                // [13][16][224]

  const int h = blockIdx.x, b = blockIdx.y;
  const int tid = threadIdx.x;
  const int w = tid >> 5, lane = tid & 31;
  const int r = lane & 15, hi = lane >> 4;

  const _Float16* gQ = Q  + (size_t)(b * H_ + h) * NP * HD;
  const _Float16* gK = K  + (size_t)(b * H_ + h) * NP * HD;
  const _Float16* gV = VT + (size_t)(b * H_ + h) * HD * NP;

  // stage K tile (vectorized): 208*64/8 = 1664 chunks
  for (int i = tid; i < NP * HD / 8; i += 416)
    ((v8h*)sK)[i] = ((const v8h*)gK)[i];
  // stage V^T with zeroed token tail 208..223: 64 rows * 28 chunks
  for (int i = tid; i < HD * (PLD / 8); i += 416) {
    const int e = i / (PLD / 8), c = i % (PLD / 8);
    v8h val = {};
    if (c < NP / 8) val = *(const v8h*)(gV + e * NP + c * 8);
    ((v8h*)sV)[i] = val;
  }
  __syncthreads();

  // q strip as two A fragments (K=64 features)
  v16h qa0 = load_frag_a(gQ + w * 16 * HD, HD);
  v16h qa1 = load_frag_a(gQ + w * 16 * HD + 32, HD);

  // ---- scores S = (q k^T) / 8 , 13 column tiles in registers ----
  const float scale = 0.125f;
  v8f S[13];
#pragma unroll
  for (int j = 0; j < 13; ++j) {
    v8f acc = {};
    v16h b0 = load_frag_b(sK + j * 16 * HD, HD);
    v16h b1 = load_frag_b(sK + j * 16 * HD + 32, HD);
    acc = WMMA16(qa0, b0, acc);
    acc = WMMA16(qa1, b1, acc);
#pragma unroll
    for (int c = 0; c < 8; ++c) S[j][c] = acc[c] * scale;
  }
  // mask invalid key columns 197..207 (tile 12, lanes with col index >= 5)
  if (r >= 5) {
#pragma unroll
    for (int c = 0; c < 8; ++c) S[12][c] = -1.0e30f;
  }

  // ---- softmax (rows live across 16-lane halves; components = row id) ----
  v8f mx = S[0];
#pragma unroll
  for (int j = 1; j < 13; ++j)
#pragma unroll
    for (int c = 0; c < 8; ++c) mx[c] = fmaxf(mx[c], S[j][c]);
#pragma unroll
  for (int c = 0; c < 8; ++c) {
    float v = mx[c];
    v = fmaxf(v, __shfl_xor(v, 1, 32));
    v = fmaxf(v, __shfl_xor(v, 2, 32));
    v = fmaxf(v, __shfl_xor(v, 4, 32));
    v = fmaxf(v, __shfl_xor(v, 8, 32));
    mx[c] = v;
  }
  v8f sum = {};
#pragma unroll
  for (int j = 0; j < 13; ++j)
#pragma unroll
    for (int c = 0; c < 8; ++c) {
      float e = __builtin_amdgcn_exp2f((S[j][c] - mx[c]) * 1.44269504f);
      S[j][c] = e;
      sum[c] += e;
    }
#pragma unroll
  for (int c = 0; c < 8; ++c) {
    float v = sum[c];
    v += __shfl_xor(v, 1, 32);
    v += __shfl_xor(v, 2, 32);
    v += __shfl_xor(v, 4, 32);
    v += __shfl_xor(v, 8, 32);
    sum[c] = __builtin_amdgcn_rcpf(v);
  }

  // normalize, spill P (f16) to this wave's LDS strip
  _Float16* Pw = sP + (size_t)w * 16 * PLD;
#pragma unroll
  for (int j = 0; j < 13; ++j)
#pragma unroll
    for (int c = 0; c < 8; ++c) {
      const float p = S[j][c] * sum[c];
      S[j][c] = p;
      Pw[(c + 8 * hi) * PLD + j * 16 + r] = (_Float16)p;
    }
  // zero K-tail columns 208..223 of P strip (16x16 halves, 8 per lane)
#pragma unroll
  for (int t = 0; t < 8; ++t) {
    const int idx = lane * 8 + t;
    Pw[(idx >> 4) * PLD + NP + (idx & 15)] = (_Float16)0.f;
  }

  // CLS-row attention stats (row 0 lives in wave 0, lanes 0..15, component 0)
  if (w == 0 && hi == 0) {
#pragma unroll
    for (int j = 0; j < 13; ++j) {
      const int col = j * 16 + r;
      if (col >= 1 && col < N_)
        ga[(size_t)(b * H_ + h) * (N_ - 1) + (col - 1)] = S[j][0];
    }
  }

  // ---- O = P @ V : 7 K-chunks of 32 tokens, 4 hd column tiles ----
  v16h pa[7];
#pragma unroll
  for (int kk = 0; kk < 7; ++kk) pa[kk] = load_frag_a(Pw + kk * 32, PLD);

#pragma unroll
  for (int n = 0; n < 4; ++n) {
    v8f acc = {};
#pragma unroll
    for (int kk = 0; kk < 7; ++kk) {
      v16h vb = load_frag_b(sV + (n * 16) * PLD + kk * 32, PLD);
      acc = WMMA16(pa[kk], vb, acc);
    }
#pragma unroll
    for (int c = 0; c < 8; ++c) {
      const int tok = w * 16 + c + 8 * hi;
      O[((size_t)b * NP + tok) * C_ + h * HD + n * 16 + r] = (_Float16)acc[c];
    }
  }
}

// ---------------------------------------------------------------------------
// Proj GEMM: out[b,n,d] = sum_c O[b,n,c] * w_proj[d,c] + b_proj[d]
// Same 64-row blocking: 1 weight fragment -> 4 WMMAs.
// ---------------------------------------------------------------------------
__global__ __launch_bounds__(256) void proj_gemm(const _Float16* __restrict__ O,
                                                 const _Float16* __restrict__ wp,
                                                 const float* __restrict__ bias,
                                                 float* __restrict__ out) {
  extern __shared__ __align__(32) char smem[];
  _Float16* sA = (_Float16*)smem;              // [64][768] = 96 KB (already f16)
  const int g   = blockIdx.x;                  // rows 64g .. 64g+63 of [B*208, 768]
  const int tid = threadIdx.x;

  const _Float16* src = O + (size_t)g * 64 * C_;
  for (int i = tid; i < 64 * C_ / 8; i += 256)
    ((v8h*)sA)[i] = ((const v8h*)src)[i];
  __syncthreads();

  const int w    = tid >> 5;
  const int lane = tid & 31;
  const int r    = lane & 15, hi = lane >> 4;

  for (int ct = 0; ct < 6; ++ct) {
    const int ctile = w * 6 + ct;              // 0..47
    const _Float16* wpt = wp + (size_t)(ctile * 16) * C_;
    __builtin_prefetch(wpt + (size_t)16 * C_, 0, 1);
    v8f acc[4] = {};
#pragma unroll 2
    for (int kc = 0; kc < 24; ++kc) {
      v16h bb = load_frag_b(wpt + kc * 32, C_);
      v16h a0 = load_frag_a(sA + 0 * 16 * C_ + kc * 32, C_);
      v16h a1 = load_frag_a(sA + 1 * 16 * C_ + kc * 32, C_);
      v16h a2 = load_frag_a(sA + 2 * 16 * C_ + kc * 32, C_);
      v16h a3 = load_frag_a(sA + 3 * 16 * C_ + kc * 32, C_);
      acc[0] = WMMA16(a0, bb, acc[0]);
      acc[1] = WMMA16(a1, bb, acc[1]);
      acc[2] = WMMA16(a2, bb, acc[2]);
      acc[3] = WMMA16(a3, bb, acc[3]);
    }
    const int d = ctile * 16 + r;
    const float bd = bias[d];
#pragma unroll
    for (int s = 0; s < 4; ++s) {
      const int m0 = g * 64 + s * 16;          // strip start row; never crosses b
      const int b  = m0 / NP;
      const int n0 = m0 - b * NP;
#pragma unroll
      for (int c = 0; c < 8; ++c) {
        const int n = n0 + c + 8 * hi;
        if (n < N_)
          out[((size_t)b * N_ + n) * C_ + d] = acc[s][c] + bd;
      }
    }
  }
}

// ---------------------------------------------------------------------------
// global_attn = mean over heads of CLS row (deterministic)
// ---------------------------------------------------------------------------
__global__ void ga_reduce(const float* __restrict__ gsc, float* __restrict__ dst) {
  const int i = blockIdx.x * blockDim.x + threadIdx.x;
  if (i < B_ * (N_ - 1)) {
    const int b = i / (N_ - 1), c = i % (N_ - 1);
    float s = 0.f;
#pragma unroll
    for (int h = 0; h < H_; ++h) s += gsc[(size_t)(b * H_ + h) * (N_ - 1) + c];
    dst[i] = s * (1.0f / H_);
  }
}

// ---------------------------------------------------------------------------
// launch
// ---------------------------------------------------------------------------
extern "C" void kernel_launch(void* const* d_in, const int* in_sizes, int n_in,
                              void* d_out, int out_size, void* d_ws, size_t ws_size,
                              hipStream_t stream) {
  const float* x      = (const float*)d_in[0];   // [64,197,768]
  const float* w_qkv  = (const float*)d_in[1];   // [2304,768]
  const float* w_proj = (const float*)d_in[2];   // [768,768]
  const float* b_proj = (const float*)d_in[3];   // [768]
  float* out = (float*)d_out;

  // carve workspace (256-byte aligned slabs)
  size_t off = 0;
  auto carve = [&](size_t bytes) -> char* {
    char* p = (char*)d_ws + off;
    off += (bytes + 255) & ~(size_t)255;
    return p;
  };
  _Float16* wqh = (_Float16*)carve((size_t)3 * C_ * C_ * 2);          // [2304,768]
  _Float16* wph = (_Float16*)carve((size_t)C_ * C_ * 2);              // [768,768]
  _Float16* Qb  = (_Float16*)carve((size_t)B_ * H_ * NP * HD * 2);    // [B,H,208,64]
  _Float16* Kb  = (_Float16*)carve((size_t)B_ * H_ * NP * HD * 2);
  _Float16* VTb = (_Float16*)carve((size_t)B_ * H_ * HD * NP * 2);    // [B,H,64,208]
  _Float16* Oh  = (_Float16*)carve((size_t)B_ * NP * C_ * 2);         // [B,208,768]
  float*    gsc = (float*)carve((size_t)B_ * H_ * (N_ - 1) * 4);      // [B,H,196]

  // weight conversion
  {
    int n1 = 3 * C_ * C_;
    cvt_f32_f16<<<(n1 + 255) / 256, 256, 0, stream>>>(w_qkv, wqh, n1);
    int n2 = C_ * C_;
    cvt_f32_f16<<<(n2 + 255) / 256, 256, 0, stream>>>(w_proj, wph, n2);
  }
  const size_t gemm_lds = (size_t)4 * 16 * C_ * 2;                    // 96 KB
  // QKV projection: 832 row tiles / 4 per WG
  qkv_gemm<<<208, 256, gemm_lds, stream>>>(x, wqh, Qb, Kb, VTb);
  // attention (148480 B dynamic LDS: K 26624 + V^T 28672 + P 93184)
  const size_t attn_lds = (size_t)(NP * HD + HD * PLD + 13 * 16 * PLD) * 2;
  attn_kernel<<<dim3(H_, B_), 416, attn_lds, stream>>>(Qb, Kb, VTb, Oh, gsc);
  // output projection
  proj_gemm<<<208, 256, gemm_lds, stream>>>(Oh, wph, b_proj, out);
  // CLS attention mean over heads -> tail of d_out
  const int nga = B_ * (N_ - 1);
  ga_reduce<<<(nga + 255) / 256, 256, 0, stream>>>(gsc, out + (size_t)B_ * N_ * C_);
}
